// Codebook_33681133535663
// MI455X (gfx1250) — matrix operand order repaired and checked by reading, not compiled, not run on hardware
//
#include <hip/hip_runtime.h>
#include <hip/hip_bf16.h>

// ---------------------------------------------------------------------------
// Cosine-sim codebook top-k + gather-sum for MI455X (gfx1250, wave32, WMMA).
//   sim = normalize(x) @ normalize(codebook)^T   via v_wmma_f32_16x16x32_bf16
//   Register blocking: 2 row-tiles x 4 col-tiles per wave (8 WMMA / 12 ds_load)
//   topk (k=32) per row, streaming (never materializes the 1 GB sim matrix)
//   out[b] = sum_j codebook[topk_idx[b][j]]
// ---------------------------------------------------------------------------

typedef __attribute__((ext_vector_type(16))) __bf16 v16bf;
typedef __attribute__((ext_vector_type(2)))  __bf16 bf16x2;
typedef __attribute__((ext_vector_type(8)))  float  v8f;

#define B_ROWS 8192
#define N_CB   32768
#define D_DIM  768
#define TOPK   32

#define BM 32                 // x rows per workgroup (2 row tiles of 16)
#define CT 4                  // col tiles (of 16) per wave
#define BN (8 * 16 * CT)      // 512 codebook rows per iteration
#define KC 128                // K chunk held in LDS

__device__ __forceinline__ __bf16 f2bf(float f) {
    unsigned u = __builtin_bit_cast(unsigned, f);
    unsigned r = (u + 0x7FFFu + ((u >> 16) & 1u)) >> 16;   // round-to-nearest-even
    unsigned short s = (unsigned short)r;
    return __builtin_bit_cast(__bf16, s);
}

// --- row L2-normalize (f32 in) -> bf16 out, one block per row --------------
__global__ void normalize_rows_kernel(const float* __restrict__ in,
                                      __bf16* __restrict__ out) {
    const int row = blockIdx.x;
    const float* r = in + (size_t)row * D_DIM;
    float ss = 0.0f;
    for (int c = threadIdx.x; c < D_DIM; c += 256) {
        float v = r[c];
        ss += v * v;
    }
    __shared__ float red[256];
    red[threadIdx.x] = ss;
    __syncthreads();
    for (int s = 128; s > 0; s >>= 1) {
        if (threadIdx.x < s) red[threadIdx.x] += red[threadIdx.x + s];
        __syncthreads();
    }
    const float inv = 1.0f / fmaxf(sqrtf(red[0]), 1e-8f);
    for (int c = threadIdx.x; c < D_DIM; c += 256)
        out[(size_t)row * D_DIM + c] = f2bf(r[c] * inv);
}

// --- fused bf16 WMMA similarity + streaming per-row top-32 -----------------
__global__ void __launch_bounds__(256)
sim_topk_kernel(const __bf16* __restrict__ xn,
                const __bf16* __restrict__ cbn,
                int* __restrict__ topk_out) {
    __shared__ __bf16 Asl[BM * D_DIM];   //  48 KB : x tile, row-major K
    __shared__ __bf16 Bsl[BN * KC];      // 128 KB : codebook chunk, row-major K
    __shared__ float  Ssl[BM * BN];      //  64 KB : sim tile
    __shared__ float  topv[BM][TOPK];    //   4 KB
    __shared__ int    topi[BM][TOPK];    //   4 KB

    const int tid  = threadIdx.x;
    const int wave = tid >> 5;
    const int lane = tid & 31;
    const int n    = lane & 15;   // M for A-frag / N for B-frag / col for C
    const int h    = lane >> 4;   // lane half
    const int row0 = blockIdx.x * BM;

    // load A tile once: 32 x 768 bf16 = 3072 x 16B
    {
        const int4* src = (const int4*)(xn + (size_t)row0 * D_DIM);
        int4* dst = (int4*)Asl;
        #pragma unroll
        for (int i = 0; i < (BM * D_DIM) / (8 * 256); ++i)
            dst[tid + i * 256] = src[tid + i * 256];
    }
    for (int i = tid; i < BM * TOPK; i += 256) {
        ((float*)topv)[i] = -3.4e38f;
        ((int*)topi)[i]   = 0;
    }
    __syncthreads();

    // per-row top-k tracking state (meaningful for tid < BM only)
    float vmin    = -3.4e38f;
    int   minslot = 0;

    for (int col0 = 0; col0 < N_CB; col0 += BN) {
        v8f acc0[CT];   // rows row0+0..15  x cols (wave*64 + c*16) .. +15
        v8f acc1[CT];   // rows row0+16..31 x same cols
        #pragma unroll
        for (int c = 0; c < CT; ++c) { acc0[c] = (v8f){}; acc1[c] = (v8f){}; }

        for (int kc = 0; kc < D_DIM; kc += KC) {
            // cooperative load: B chunk = cbn[col0..col0+511][kc..kc+127]
            {
                int4* dst = (int4*)Bsl;
                #pragma unroll
                for (int i = 0; i < (BN * KC) / (8 * 256); ++i) {
                    int chunk = tid + i * 256;        // 16B each
                    int rr = chunk >> 4;              // local codebook row
                    int kb = (chunk & 15) << 3;       // bf16 offset in chunk
                    dst[chunk] = *(const int4*)(cbn + (size_t)(col0 + rr) * D_DIM + kc + kb);
                }
            }
            __syncthreads();

            #pragma unroll
            for (int ks = 0; ks < KC; ks += 32) {
                // A fragments: lane = M, VGPR v -> K = (v<4 ? 2v+8h : 8+2v+8h)
                v16bf a0, a1;
                #pragma unroll
                for (int v = 0; v < 8; ++v) {
                    int kk = (v < 4) ? (2 * v + 8 * h) : (8 + 2 * v + 8 * h);
                    bf16x2 p0 = *(const bf16x2*)&Asl[(size_t)n * D_DIM + kc + ks + kk];
                    bf16x2 p1 = *(const bf16x2*)&Asl[(size_t)(n + 16) * D_DIM + kc + ks + kk];
                    a0[2 * v] = p0[0]; a0[2 * v + 1] = p0[1];
                    a1[2 * v] = p1[0]; a1[2 * v + 1] = p1[1];
                }
                // B fragments: lane = N (codebook col), VGPR v -> K = (v<4 ? 2v+8h : 8+2v+8h)
                v16bf bfrag[CT];
                #pragma unroll
                for (int c = 0; c < CT; ++c) {
                    #pragma unroll
                    for (int v = 0; v < 8; ++v) {
                        int kk = (v < 4) ? (2 * v + 8 * h) : (8 + 2 * v + 8 * h);
                        bf16x2 p = *(const bf16x2*)
                            &Bsl[(size_t)(wave * (16 * CT) + c * 16 + n) * KC + ks + kk];
                        bfrag[c][2 * v]     = p[0];
                        bfrag[c][2 * v + 1] = p[1];
                    }
                }
                #pragma unroll
                for (int c = 0; c < CT; ++c) {
                    acc0[c] = __builtin_amdgcn_wmma_f32_16x16x32_bf16(
                                  false, a0, false, bfrag[c], (short)0, acc0[c], false, false);
                    acc1[c] = __builtin_amdgcn_wmma_f32_16x16x32_bf16(
                                  false, a1, false, bfrag[c], (short)0, acc1[c], false, false);
                }
            }
            __syncthreads();
        }

        // C/D layout: VGPR v, lane half h -> row v + 8h, col = lane&15
        #pragma unroll
        for (int c = 0; c < CT; ++c) {
            const int colb = wave * (16 * CT) + c * 16 + n;
            #pragma unroll
            for (int v = 0; v < 8; ++v) {
                Ssl[(v + 8 * h) * BN + colb]      = acc0[c][v];
                Ssl[(16 + v + 8 * h) * BN + colb] = acc1[c][v];
            }
        }
        __syncthreads();

        // streaming top-32: one lane per x-row scans 512 new candidates
        if (tid < BM) {
            const float* srow = &Ssl[tid * BN];
            for (int c = 0; c < BN; ++c) {
                float s = srow[c];
                if (s > vmin) {
                    topv[tid][minslot] = s;
                    topi[tid][minslot] = col0 + c;
                    float nv = topv[tid][0];
                    int   ns = 0;
                    #pragma unroll
                    for (int j = 1; j < TOPK; ++j) {
                        float tv = topv[tid][j];
                        if (tv < nv) { nv = tv; ns = j; }
                    }
                    vmin = nv;
                    minslot = ns;
                }
            }
        }
        __syncthreads();
    }

    if (tid < BM) {
        #pragma unroll
        for (int j = 0; j < TOPK; ++j)
            topk_out[(size_t)(row0 + tid) * TOPK + j] = topi[tid][j];
    }
}

// --- gather top-32 original codebook rows and sum, one block per x-row -----
__global__ void gather_sum_kernel(const float* __restrict__ cb,
                                  const int* __restrict__ topk,
                                  float* __restrict__ out) {
    const int row = blockIdx.x;
    __shared__ int sidx[TOPK];
    if (threadIdx.x < TOPK) sidx[threadIdx.x] = topk[(size_t)row * TOPK + threadIdx.x];
    __syncthreads();
    for (int c = threadIdx.x; c < D_DIM; c += 256) {
        float acc = 0.0f;
        #pragma unroll
        for (int j = 0; j < TOPK; ++j)
            acc += cb[(size_t)sidx[j] * D_DIM + c];
        out[(size_t)row * D_DIM + c] = acc;
    }
}

extern "C" void kernel_launch(void* const* d_in, const int* in_sizes, int n_in,
                              void* d_out, int out_size, void* d_ws, size_t ws_size,
                              hipStream_t stream) {
    const float* x  = (const float*)d_in[0];   // 8192 x 768
    const float* cb = (const float*)d_in[1];   // 32768 x 768
    (void)in_sizes; (void)n_in; (void)out_size; (void)ws_size;

    // workspace layout (all 16B aligned)
    char* ws = (char*)d_ws;
    __bf16* xbf  = (__bf16*)ws;                                   // 12,582,912 B
    __bf16* cbbf = (__bf16*)(ws + (size_t)B_ROWS * D_DIM * 2);    // 50,331,648 B
    int*    topk = (int*)(ws + (size_t)B_ROWS * D_DIM * 2
                             + (size_t)N_CB * D_DIM * 2);         //  1,048,576 B

    normalize_rows_kernel<<<B_ROWS, 256, 0, stream>>>(x,  xbf);
    normalize_rows_kernel<<<N_CB,   256, 0, stream>>>(cb, cbbf);
    sim_topk_kernel<<<B_ROWS / BM, 256, 0, stream>>>(xbf, cbbf, topk);
    gather_sum_kernel<<<B_ROWS, 256, 0, stream>>>(cb, topk, (float*)d_out);
}
